// LigandGraphEncoder_91164975824918
// MI455X (gfx1250) — compile-verified, matrix operand order checked by reading, not run on hardware
//
#include <hip/hip_runtime.h>
#include <hip/hip_bf16.h>

typedef __attribute__((ext_vector_type(16))) __bf16 v16bf;
typedef __attribute__((ext_vector_type(8)))  __bf16 v8bf;
typedef __attribute__((ext_vector_type(8)))  float  v8f;

#define N_NODES 20000
#define N_EDGES 320000
#define EDGE_TOT (N_EDGES + N_NODES)   // 340000, incl. self-loops
#define EMB 32
#define HID 128
#define NH 4
#define OUTD (NH * HID)                // 512
#define NB 64
#define NEG_SLOPE 0.2f
#define LN_EPS 1e-5f

static __host__ int cdiv(long long a, long long b) { return (int)((a + b - 1) / b); }

// ---- order-preserving float <-> uint encoding for atomicMax on floats ----
static __device__ __forceinline__ unsigned enc_f(float f) {
  unsigned b = __float_as_uint(f);
  return (b & 0x80000000u) ? ~b : (b | 0x80000000u);
}
static __device__ __forceinline__ float dec_f(unsigned u) {
  unsigned b = (u & 0x80000000u) ? (u ^ 0x80000000u) : ~u;
  return __uint_as_float(b);
}
#define ENC_NEG_INF 0x007FFFFFu   // enc_f(-inf)

// -------------------- utility fills --------------------
__global__ void lge_fill_f32(float* p, float v, int n) {
  int i = blockIdx.x * blockDim.x + threadIdx.x;
  if (i < n) p[i] = v;
}
__global__ void lge_fill_u32(unsigned* p, unsigned v, int n) {
  int i = blockIdx.x * blockDim.x + threadIdx.x;
  if (i < n) p[i] = v;
}
__global__ void lge_f32_to_bf16(const float* __restrict__ s, __bf16* __restrict__ d, int n) {
  int i = blockIdx.x * blockDim.x + threadIdx.x;
  if (i < n) d[i] = (__bf16)s[i];
}

// -------------------- self-loop attr (PyG fill_value='mean') --------------------
__global__ void lge_deg_loop(const float* __restrict__ eattr, const int* __restrict__ ei,
                             float* __restrict__ deg, float* __restrict__ loopb) {
  int gid = blockIdx.x * blockDim.x + threadIdx.x;
  if (gid >= N_EDGES * EMB) return;
  int e = gid >> 5;      // /EMB
  int k = gid & 31;
  int dst = ei[N_EDGES + e];
  atomicAdd(&loopb[(size_t)dst * EMB + k], eattr[gid]);
  if (k == 0) atomicAdd(&deg[dst], 1.0f);
}
__global__ void lge_loop_div(float* __restrict__ loopb, const float* __restrict__ deg) {
  int gid = blockIdx.x * blockDim.x + threadIdx.x;
  if (gid >= N_NODES * EMB) return;
  loopb[gid] /= fmaxf(deg[gid >> 5], 1.0f);
}

// -------------------- combined attention weights: comb[k][h] = sum_c W[k, h*HID+c]*att[h,c] --------------------
__global__ void lge_comb(const float* __restrict__ W, const float* __restrict__ att,
                         float* __restrict__ comb, int K) {
  int gid = blockIdx.x * blockDim.x + threadIdx.x;
  if (gid >= K * NH) return;
  int k = gid / NH, h = gid % NH;
  float s = 0.f;
  for (int c = 0; c < HID; ++c) s += W[(size_t)k * OUTD + h * HID + c] * att[h * HID + c];
  comb[gid] = s;
}

// -------------------- W^T -> bf16 so B fragments load contiguously along K --------------------
__global__ void lge_wt_bf16(const float* __restrict__ W, __bf16* __restrict__ Wt, int K) {
  int gid = blockIdx.x * blockDim.x + threadIdx.x;
  if (gid >= OUTD * K) return;
  int n = gid / K, k = gid % K;
  Wt[gid] = (__bf16)W[(size_t)k * OUTD + n];
}

// -------------------- WMMA GEMM: C[N_NODES,512] = A(bf16)[N_NODES,K] x Bt(bf16)[512,K]^T --------------------
// One wave per 16x16 C tile; K in {32,128}; K-step of 32 via v_wmma_f32_16x16x32_bf16.
// 16-bit A layout (ISA 7.12.2): lanes 0-15 row M=lane, K in {0..7,16..23}; lanes 16-31 K in {8..15,24..31}.
// B (column n of B == row n of Bt) mirrors the A per-lane layout.
__global__ void lge_gemm_bf16_wmma(const __bf16* __restrict__ A, const __bf16* __restrict__ Bt,
                                   float* __restrict__ C, int K) {
  int wave = blockIdx.x * (blockDim.x >> 5) + (threadIdx.x >> 5);
  int lane = threadIdx.x & 31;
  int tn = wave & 31;          // 32 tiles over 512 cols
  int tm = wave >> 5;          // 1250 tiles over 20000 rows
  if (tm >= N_NODES / 16) return;
  int r16  = lane & 15;
  int row  = tm * 16 + r16;
  int col  = tn * 16 + r16;
  int koff = (lane < 16) ? 0 : 8;
  v8f acc = {};
  for (int kb = 0; kb < K; kb += 32) {
    const __bf16* ap = A  + (size_t)row * K + kb + koff;
    const __bf16* bp = Bt + (size_t)col * K + kb + koff;
    v8bf a0 = *(const v8bf*)ap;
    v8bf a1 = *(const v8bf*)(ap + 16);
    v8bf b0 = *(const v8bf*)bp;
    v8bf b1 = *(const v8bf*)(bp + 16);
    v16bf a, b;
#pragma unroll
    for (int i = 0; i < 8; ++i) { a[i] = a0[i]; a[i + 8] = a1[i]; b[i] = b0[i]; b[i + 8] = b1[i]; }
    acc = __builtin_amdgcn_wmma_f32_16x16x32_bf16(false, a, false, b, (short)0, acc, false, false);
  }
  int rbase = tm * 16 + ((lane < 16) ? 0 : 8);
#pragma unroll
  for (int r = 0; r < 8; ++r) C[(size_t)(rbase + r) * OUTD + col] = acc[r];
}

// -------------------- per-node attention logits: out[n][h] = x[n,:] . comb[:,h] --------------------
__global__ void lge_alpha_node(const float* __restrict__ x, const float* __restrict__ comb,
                               float* __restrict__ out, int K) {
  int gid = blockIdx.x * blockDim.x + threadIdx.x;
  if (gid >= N_NODES * NH) return;
  int n = gid / NH, h = gid % NH;
  float s = 0.f;
  for (int k = 0; k < K; ++k) s += x[(size_t)n * K + k] * comb[k * NH + h];
  out[gid] = s;
}

// -------------------- per-edge attention logits (incl. self-loops with mean attr) --------------------
__global__ void lge_alpha_edge(const float* __restrict__ eattr, const float* __restrict__ loopb,
                               const float* __restrict__ ce, float* __restrict__ ae) {
  int gid = blockIdx.x * blockDim.x + threadIdx.x;
  if (gid >= EDGE_TOT * NH) return;
  int e = gid / NH, h = gid % NH;
  const float* row = (e < N_EDGES) ? (eattr + (size_t)e * EMB)
                                   : (loopb + (size_t)(e - N_EDGES) * EMB);
  float s = 0.f;
  for (int k = 0; k < EMB; ++k) s += row[k] * ce[k * NH + h];
  ae[gid] = s;
}

// -------------------- segment softmax pass 1: leaky-relu logits + per-(dst,h) max --------------------
__global__ void lge_logit_max(const float* __restrict__ as, const float* __restrict__ ad,
                              const float* __restrict__ ae, const int* __restrict__ ei,
                              float* __restrict__ logit, unsigned* __restrict__ mbuf) {
  int gid = blockIdx.x * blockDim.x + threadIdx.x;
  if (gid >= EDGE_TOT * NH) return;
  int e = gid / NH, h = gid % NH;
  int s, d;
  if (e < N_EDGES) { s = ei[e]; d = ei[N_EDGES + e]; } else { s = d = e - N_EDGES; }
  float al = as[s * NH + h] + ad[d * NH + h] + ae[gid];
  al = (al > 0.f) ? al : NEG_SLOPE * al;
  logit[gid] = al;
  atomicMax(&mbuf[d * NH + h], enc_f(al));
}

// -------------------- pass 2: w = exp(al - m); sum per (dst,h); w overwrites logit in place --------------------
__global__ void lge_expw_sum(float* __restrict__ logit, const unsigned* __restrict__ mbuf,
                             float* __restrict__ sbuf, const int* __restrict__ ei) {
  int gid = blockIdx.x * blockDim.x + threadIdx.x;
  if (gid >= EDGE_TOT * NH) return;
  int e = gid / NH, h = gid % NH;
  int d = (e < N_EDGES) ? ei[N_EDGES + e] : (e - N_EDGES);
  float m = dec_f(mbuf[d * NH + h]);
  float w = __expf(logit[gid] - m);
  logit[gid] = w;
  atomicAdd(&sbuf[d * NH + h], w);
}

// -------------------- pass 3: one wave per (edge,head), scatter coeff * h[src] into acc[dst] --------------------
__global__ void lge_scatter(const float* __restrict__ hbuf, const float* __restrict__ wbuf,
                            const float* __restrict__ sbuf, const int* __restrict__ ei,
                            float* __restrict__ acc) {
  int warp = (blockIdx.x * blockDim.x + threadIdx.x) >> 5;
  int lane = threadIdx.x & 31;
  if (warp >= EDGE_TOT * NH) return;
  int e = warp >> 2;           // /NH
  int h = warp & 3;
  int s, d;
  if (e < N_EDGES) { s = ei[e]; d = ei[N_EDGES + e]; } else { s = d = e - N_EDGES; }
  float coeff = wbuf[warp] / sbuf[d * NH + h];
  const float4* hp = (const float4*)(hbuf + (size_t)s * OUTD + h * HID);
  float4 v = hp[lane];
  float* ap = acc + (size_t)d * OUTD + h * HID + lane * 4;
  atomicAdd(ap + 0, coeff * v.x);
  atomicAdd(ap + 1, coeff * v.y);
  atomicAdd(ap + 2, coeff * v.z);
  atomicAdd(ap + 3, coeff * v.w);
}

// -------------------- head-mean + bias + LayerNorm + ReLU, one block (128 threads) per node --------------------
__global__ void lge_finalize(const float* __restrict__ acc, const float* __restrict__ bias,
                             const float* __restrict__ g, const float* __restrict__ bb,
                             float* __restrict__ xf) {
  int n = blockIdx.x;
  int c = threadIdx.x;
  const float* a = acc + (size_t)n * OUTD;
  float v = (a[c] + a[HID + c] + a[2 * HID + c] + a[3 * HID + c]) * 0.25f + bias[c];
  __shared__ float red[HID];
  red[c] = v; __syncthreads();
  for (int st = HID / 2; st > 0; st >>= 1) { if (c < st) red[c] += red[c + st]; __syncthreads(); }
  float mu = red[0] * (1.0f / HID);
  __syncthreads();
  float dlt = v - mu;
  red[c] = dlt * dlt; __syncthreads();
  for (int st = HID / 2; st > 0; st >>= 1) { if (c < st) red[c] += red[c + st]; __syncthreads(); }
  float var = red[0] * (1.0f / HID);
  float y = dlt * rsqrtf(var + LN_EPS) * g[c] + bb[c];
  xf[(size_t)n * HID + c] = fmaxf(y, 0.f);
}

// -------------------- global mean pool over graphs --------------------
__global__ void lge_pool_add(const float* __restrict__ xf, const int* __restrict__ batch,
                             float* __restrict__ pool, float* __restrict__ cnt) {
  int gid = blockIdx.x * blockDim.x + threadIdx.x;
  if (gid >= N_NODES * HID) return;
  int n = gid >> 7, c = gid & 127;
  int b = batch[n];
  atomicAdd(&pool[b * HID + c], xf[gid]);
  if (c == 0) atomicAdd(&cnt[b], 1.f);
}
__global__ void lge_pool_div(const float* __restrict__ pool, const float* __restrict__ cnt,
                             float* __restrict__ out) {
  int gid = blockIdx.x * blockDim.x + threadIdx.x;
  if (gid >= NB * HID) return;
  out[gid] = pool[gid] / fmaxf(cnt[gid >> 7], 1.f);
}

extern "C" void kernel_launch(void* const* d_in, const int* in_sizes, int n_in,
                              void* d_out, int out_size, void* d_ws, size_t ws_size,
                              hipStream_t stream) {
  const float* x_in     = (const float*)d_in[0];
  const float* eattr    = (const float*)d_in[1];
  const float* lin0     = (const float*)d_in[2];
  const float* lin1     = (const float*)d_in[3];
  const float* lin2     = (const float*)d_in[4];
  const float* lin_edge = (const float*)d_in[5];
  const float* att_src  = (const float*)d_in[6];
  const float* att_dst  = (const float*)d_in[7];
  const float* att_edge = (const float*)d_in[8];
  const float* bias     = (const float*)d_in[9];
  const float* ln_g     = (const float*)d_in[10];
  const float* ln_b     = (const float*)d_in[11];
  const int*   ei       = (const int*)d_in[12];   // [2, E] flat: src then dst
  const int*   batch    = (const int*)d_in[13];

  // ---- workspace partition (256B-aligned) ----
  char* base = (char*)d_ws;
  size_t off = 0;
  auto take = [&](size_t bytes) -> void* {
    void* p = base + off;
    off += (bytes + 255) & ~(size_t)255;
    return p;
  };
  float*    deg   = (float*)take((size_t)N_NODES * 4);
  float*    loopb = (float*)take((size_t)N_NODES * EMB * 4);
  float*    xf    = (float*)take((size_t)N_NODES * HID * 4);
  __bf16*   xb    = (__bf16*)take((size_t)N_NODES * HID * 2);
  __bf16*   wtb   = (__bf16*)take((size_t)OUTD * HID * 2);
  float*    hbuf  = (float*)take((size_t)N_NODES * OUTD * 4);
  float*    accb  = (float*)take((size_t)N_NODES * OUTD * 4);
  float*    asrcn = (float*)take((size_t)N_NODES * NH * 4);
  float*    adstn = (float*)take((size_t)N_NODES * NH * 4);
  float*    aee   = (float*)take((size_t)EDGE_TOT * NH * 4);
  float*    logit = (float*)take((size_t)EDGE_TOT * NH * 4);
  unsigned* mbuf  = (unsigned*)take((size_t)N_NODES * NH * 4);
  float*    sbuf  = (float*)take((size_t)N_NODES * NH * 4);
  float*    csrc  = (float*)take((size_t)HID * NH * 4);
  float*    cdst  = (float*)take((size_t)HID * NH * 4);
  float*    ce    = (float*)take((size_t)EMB * NH * 4);
  float*    pool  = (float*)take((size_t)NB * HID * 4);
  float*    cnt   = (float*)take((size_t)NB * 4);
  (void)ws_size; (void)n_in; (void)in_sizes; (void)out_size;

  // ---- self-loop edge attributes (mean of incoming) ----
  lge_fill_f32<<<cdiv(N_NODES, 256), 256, 0, stream>>>(deg, 0.f, N_NODES);
  lge_fill_f32<<<cdiv((long long)N_NODES * EMB, 256), 256, 0, stream>>>(loopb, 0.f, N_NODES * EMB);
  lge_deg_loop<<<cdiv((long long)N_EDGES * EMB, 256), 256, 0, stream>>>(eattr, ei, deg, loopb);
  lge_loop_div<<<cdiv((long long)N_NODES * EMB, 256), 256, 0, stream>>>(loopb, deg);

  const float* Ws[3] = {lin0, lin1, lin2};
  const float* xcur = x_in;
  int K = EMB;

  for (int l = 0; l < 3; ++l) {
    const float* W = Ws[l];
    // combined attention weight vectors (collapses the big edge GEMM + h-dots into GEMVs)
    lge_comb<<<cdiv(K * NH, 128), 128, 0, stream>>>(W, att_src + l * NH * HID, csrc, K);
    lge_comb<<<cdiv(K * NH, 128), 128, 0, stream>>>(W, att_dst + l * NH * HID, cdst, K);
    lge_comb<<<cdiv(EMB * NH, 128), 128, 0, stream>>>(lin_edge + (size_t)l * EMB * OUTD,
                                                      att_edge + l * NH * HID, ce, EMB);
    // bf16 operands for WMMA
    lge_wt_bf16<<<cdiv((long long)OUTD * K, 256), 256, 0, stream>>>(W, wtb, K);
    lge_f32_to_bf16<<<cdiv((long long)N_NODES * K, 256), 256, 0, stream>>>(xcur, xb, N_NODES * K);
    // h = x @ W via v_wmma_f32_16x16x32_bf16 (1250 x 32 tiles, 4 waves/block)
    int tiles = (N_NODES / 16) * (OUTD / 16);
    lge_gemm_bf16_wmma<<<tiles / 4, 128, 0, stream>>>(xb, wtb, hbuf, K);
    // attention logits
    lge_alpha_node<<<cdiv((long long)N_NODES * NH, 256), 256, 0, stream>>>(xcur, csrc, asrcn, K);
    lge_alpha_node<<<cdiv((long long)N_NODES * NH, 256), 256, 0, stream>>>(xcur, cdst, adstn, K);
    lge_alpha_edge<<<cdiv((long long)EDGE_TOT * NH, 256), 256, 0, stream>>>(eattr, loopb, ce, aee);
    // segment softmax + scatter
    lge_fill_u32<<<cdiv((long long)N_NODES * NH, 256), 256, 0, stream>>>(mbuf, ENC_NEG_INF, N_NODES * NH);
    lge_fill_f32<<<cdiv((long long)N_NODES * NH, 256), 256, 0, stream>>>(sbuf, 0.f, N_NODES * NH);
    lge_fill_f32<<<cdiv((long long)N_NODES * OUTD, 256), 256, 0, stream>>>(accb, 0.f, N_NODES * OUTD);
    lge_logit_max<<<cdiv((long long)EDGE_TOT * NH, 256), 256, 0, stream>>>(asrcn, adstn, aee, ei, logit, mbuf);
    lge_expw_sum<<<cdiv((long long)EDGE_TOT * NH, 256), 256, 0, stream>>>(logit, mbuf, sbuf, ei);
    lge_scatter<<<cdiv((long long)EDGE_TOT * NH * 32, 256), 256, 0, stream>>>(hbuf, logit, sbuf, ei, accb);
    // head mean + bias + LayerNorm + ReLU
    lge_finalize<<<N_NODES, HID, 0, stream>>>(accb, bias + l * HID, ln_g + l * HID, ln_b + l * HID, xf);
    xcur = xf;
    K = HID;
  }

  // global mean pool
  lge_fill_f32<<<cdiv((long long)NB * HID, 256), 256, 0, stream>>>(pool, 0.f, NB * HID);
  lge_fill_f32<<<cdiv(NB, 256), 256, 0, stream>>>(cnt, 0.f, NB);
  lge_pool_add<<<cdiv((long long)N_NODES * HID, 256), 256, 0, stream>>>(xf, batch, pool, cnt);
  lge_pool_div<<<cdiv((long long)NB * HID, 256), 256, 0, stream>>>(pool, cnt, (float*)d_out);
}
